// Net_25563645345835
// MI455X (gfx1250) — matrix operand order repaired and checked by reading, not compiled, not run on hardware
//
#include <hip/hip_runtime.h>
#include <hip/hip_bf16.h>

// ---------------- problem constants (match reference) ----------------
#define cN1 100000
#define cE1 1000000
#define cN2 200000
#define cE2 2000000
#define cN3 300000
#define cE3 3000000
#define cA2 400000
#define cA3 900000
#define cG  256
#define cF0 16
#define cNI2 12
#define cNI3 24
#define cC  10

typedef __bf16 v16bf __attribute__((ext_vector_type(16)));
typedef __bf16 v2bf  __attribute__((ext_vector_type(2)));
typedef float  v8f   __attribute__((ext_vector_type(8)));

__device__ __forceinline__ float eluf(float x) { return x > 0.f ? x : expm1f(x); }

// ---------------- utility kernels ----------------
__global__ void k_zero(float* __restrict__ p, long long n) {
    long long i = (long long)blockIdx.x * blockDim.x + threadIdx.x;
    if (i < n) p[i] = 0.f;
}

// agg[dst] += x[src] over edges; one thread per (edge, 4-feature group)
__global__ void k_scatter_edges(const float* __restrict__ x, const int* __restrict__ ei,
                                float* __restrict__ agg, int E, int F4, int F) {
    long long i = (long long)blockIdx.x * blockDim.x + threadIdx.x;
    if (i >= (long long)E * F4) return;
    int e = (int)(i / F4);
    int g = (int)(i % F4) * 4;
    int s = ei[e];
    int d = ei[E + e];
    const float4 v = *(const float4*)(x + (long long)s * F + g);
    float* o = agg + (long long)d * F + g;
    atomicAdd(o + 0, v.x); atomicAdd(o + 1, v.y);
    atomicAdd(o + 2, v.z); atomicAdd(o + 3, v.w);
}

// sum[dst] += h[src] (64-wide), cnt[dst] += 1, over assignment pairs
__global__ void k_scatter_assign(const float* __restrict__ h, const int* __restrict__ src,
                                 const int* __restrict__ dst, float* __restrict__ sum,
                                 float* __restrict__ cnt, int A) {
    long long i = (long long)blockIdx.x * blockDim.x + threadIdx.x;
    if (i >= (long long)A * 16) return;
    int a = (int)(i / 16);
    int g = (int)(i % 16) * 4;
    int s = src[a];
    int d = dst[a];
    const float4 v = *(const float4*)(h + (long long)s * 64 + g);
    float* o = sum + (long long)d * 64 + g;
    atomicAdd(o + 0, v.x); atomicAdd(o + 1, v.y);
    atomicAdd(o + 2, v.z); atomicAdd(o + 3, v.w);
    if (g == 0) atomicAdd(cnt + d, 1.f);
}

// out[:, :64] = sum/max(cnt,1), out[:, 64:64+ni] = iso
__global__ void k_build_tuple(const float* __restrict__ sum, const float* __restrict__ cnt,
                              const float* __restrict__ iso, float* __restrict__ out,
                              int n, int ni) {
    const int F = 64 + ni;
    long long i = (long long)blockIdx.x * blockDim.x + threadIdx.x;
    if (i >= (long long)n * F) return;
    int r = (int)(i / F);
    int c = (int)(i % F);
    out[i] = (c < 64) ? sum[(long long)r * 64 + c] / fmaxf(cnt[r], 1.f)
                      : iso[(long long)r * ni + (c - 64)];
}

// batch-mean scatter into xcat[G,192] at column offset
__global__ void k_pool_scatter(const float* __restrict__ h, const int* __restrict__ batch,
                               float* __restrict__ xcat, float* __restrict__ gc,
                               int n, int colofs) {
    long long i = (long long)blockIdx.x * blockDim.x + threadIdx.x;
    if (i >= (long long)n * 16) return;
    int r = (int)(i / 16);
    int g = (int)(i % 16) * 4;
    int b = batch[r];
    const float4 v = *(const float4*)(h + (long long)r * 64 + g);
    float* o = xcat + (long long)b * 192 + colofs + g;
    atomicAdd(o + 0, v.x); atomicAdd(o + 1, v.y);
    atomicAdd(o + 2, v.z); atomicAdd(o + 3, v.w);
    if (g == 0) atomicAdd(gc + b, 1.f);
}

__global__ void k_pool_div(float* __restrict__ xcat, const float* __restrict__ gc, int colofs) {
    int i = blockIdx.x * blockDim.x + threadIdx.x;
    if (i >= cG * 64) return;
    int b = i / 64, c = i % 64;
    xcat[(long long)b * 192 + colofs + c] /= fmaxf(gc[b], 1.f);
}

// ---------------- WMMA GraphConv GEMM ----------------
// OUT[n,fo] = elu( X[n,K] @ Wr[K,fo] + AGG[n,K] @ Wn[K,fo] + bias )
// bf16 operands, f32 accumulate. One wave: 16 rows x (NT*16) cols; 8 waves = 128 rows/block.
//  * Weight fragments: staged once per block into LDS in per-lane B-fragment layout
//    (cooperative, __syncthreads once), read as 32B ds loads in the K-loop.
//  * A fragments: each wave loads its 16x32 f32 tile with coalesced float2 reads
//    (each 16-lane half covers one contiguous 128B row), converts to bf16 once,
//    stores pairs into wave-private LDS in A-fragment layout (32 distinct banks),
//    then reads one 32B fragment per matrix. No barrier: LDS is in-order per wave.
#define MAX_KSTEPS 3   // K <= 96 for all layers here

template <int NT>
__global__ void __launch_bounds__(256)
k_gconv_wmma(const float* __restrict__ X, const float* __restrict__ AGG,
             const float* __restrict__ Wr, const float* __restrict__ Wn,
             const float* __restrict__ bias, float* __restrict__ OUT,
             int n, int K, int do_elu) {
    const int fo   = NT * 16;
    const int tid  = threadIdx.x;
    const int lane = tid & 31;
    const int wave = tid >> 5;
    const int half = lane >> 4;      // 0: lanes 0-15, 1: lanes 16-31
    const int m    = lane & 15;
    const int row0 = blockIdx.x * 128 + wave * 16;
    const int ksteps = (K + 31) >> 5;

    __shared__ __bf16 wf[MAX_KSTEPS * 2 * 4 * 32 * 16];  // [ks][mat][t][lane][16]
    __shared__ __bf16 af[8][2][32 * 16];                 // [wave][mat][lane][16]

    // ---- cooperative weight staging: f32 -> bf16 pairs in B-fragment layout ----
    {
        const int total = ksteps * 2 * NT * 32 * 8;   // bf16 pairs
        for (int p = tid; p < total; p += 256) {
            int v    = p & 7;
            int ln   = (p >> 3) & 31;
            int t    = (p >> 8) % NT;
            int mat  = (p / (256 * NT)) & 1;
            int ks   = p / (512 * NT);
            int hh   = ln >> 4;
            int col  = t * 16 + (ln & 15);
            int k    = (ks << 5) + ((v >> 2) << 4) + (hh << 3) + ((v & 3) << 1);   // lanes0-15: K0..7,16..23, lanes16-31: K8..15,24..31
            const float* W = mat ? Wn : Wr;
            float w0 = (k < K) ? W[(long long)k * fo + col] : 0.f;          // K even
            float w1 = (k < K) ? W[(long long)(k + 1) * fo + col] : 0.f;
            int base = ((((ks * 2 + mat) * NT + t) * 32 + ln) << 4) + (v << 1);
            wf[base]     = (__bf16)w0;
            wf[base + 1] = (__bf16)w1;
        }
    }
    __syncthreads();

    v8f acc[NT];
#pragma unroll
    for (int t = 0; t < NT; t++) {
        float bv = bias[t * 16 + m];
#pragma unroll
        for (int j = 0; j < 8; j++) acc[t][j] = bv;
    }

    for (int ks = 0; ks < ksteps; ks++) {
        const int k0 = ks << 5;
        // ---- stage this wave's 16x32 A tiles (X and AGG), coalesced ----
#pragma unroll
        for (int mat = 0; mat < 2; mat++) {
            const float* S = mat ? AGG : X;
#pragma unroll
            for (int i = 0; i < 8; i++) {
                int p    = i * 32 + lane;     // 0..255 pair slots
                int r    = p >> 4;            // tile row 0..15
                int pi   = p & 15;            // k-pair index 0..15
                int k    = k0 + (pi << 1);
                int grow = row0 + r;
                float2 val = (grow < n && k < K)
                                 ? *(const float2*)(S + (long long)grow * K + k)
                                 : make_float2(0.f, 0.f);
                // fragment coords: kb=2*pi -> half=(pi>>2)&1, v=((pi>>3)<<2)+(pi&3)
                int hh = (pi >> 2) & 1;
                int vv = ((pi >> 3) << 2) + (pi & 3);
                int dl = hh * 16 + r;
                v2bf pk; pk[0] = (__bf16)val.x; pk[1] = (__bf16)val.y;
                *(v2bf*)&af[wave][mat][(dl << 4) + (vv << 1)] = pk;
            }
        }
        const v16bf aX = *(const v16bf*)&af[wave][0][lane << 4];
        const v16bf aA = *(const v16bf*)&af[wave][1][lane << 4];

#pragma unroll
        for (int t = 0; t < NT; t++) {
            const v16bf bR = *(const v16bf*)&wf[(((ks * 2 + 0) * NT + t) * 32 + lane) << 4];
            const v16bf bN = *(const v16bf*)&wf[(((ks * 2 + 1) * NT + t) * 32 + lane) << 4];
            acc[t] = __builtin_amdgcn_wmma_f32_16x16x32_bf16(false, aX, false, bR,
                                                             (short)0, acc[t], false, false);
            acc[t] = __builtin_amdgcn_wmma_f32_16x16x32_bf16(false, aA, false, bN,
                                                             (short)0, acc[t], false, false);
        }
    }

    // ---- store (C/D layout: VGPR j -> row j (lanes0-15) / 8+j (lanes16-31)) ----
#pragma unroll
    for (int t = 0; t < NT; t++) {
        int col = t * 16 + m;
#pragma unroll
        for (int j = 0; j < 8; j++) {
            int r = row0 + half * 8 + j;
            if (r < n) {
                float v = acc[t][j];
                OUT[(long long)r * fo + col] = do_elu ? eluf(v) : v;
            }
        }
    }
}

// ---------------- FC head ----------------
__global__ void k_fc(const float* __restrict__ in, const float* __restrict__ W,
                     const float* __restrict__ b, float* __restrict__ out,
                     int n, int K, int fo, int do_elu) {
    int i = blockIdx.x * blockDim.x + threadIdx.x;
    if (i >= n * fo) return;
    int r = i / fo, o = i % fo;
    float s = b[o];
    for (int k = 0; k < K; k++) s += in[(long long)r * K + k] * W[(long long)k * fo + o];
    out[i] = do_elu ? eluf(s) : s;
}

__global__ void k_logsoftmax(const float* __restrict__ z, float* __restrict__ out, int n) {
    int r = blockIdx.x * blockDim.x + threadIdx.x;
    if (r >= n) return;
    float mx = -1e30f;
    for (int c = 0; c < cC; c++) mx = fmaxf(mx, z[r * cC + c]);
    float s = 0.f;
    for (int c = 0; c < cC; c++) s += __expf(z[r * cC + c] - mx);
    float l = mx + logf(s);
    for (int c = 0; c < cC; c++) out[r * cC + c] = z[r * cC + c] - l;
}

// ---------------- host orchestration ----------------
static inline int cdiv_i(long long a, int b) { return (int)((a + b - 1) / b); }

extern "C" void kernel_launch(void* const* d_in, const int* in_sizes, int n_in,
                              void* d_out, int out_size, void* d_ws, size_t ws_size,
                              hipStream_t stream) {
    const float* x    = (const float*)d_in[0];
    const float* iso2 = (const float*)d_in[1];
    const float* iso3 = (const float*)d_in[2];
    const int* ei1 = (const int*)d_in[3];
    const int* ei2 = (const int*)d_in[4];
    const int* ei3 = (const int*)d_in[5];
    const int* as2s = (const int*)d_in[6];
    const int* as2d = (const int*)d_in[7];
    const int* as3s = (const int*)d_in[8];
    const int* as3d = (const int*)d_in[9];
    const int* b1 = (const int*)d_in[10];
    const int* b2 = (const int*)d_in[11];
    const int* b3 = (const int*)d_in[12];
    // params (dict order): c1..c7 (wr,wn,b) then fc1..fc3 (w,b)
    const float* P[27];
    for (int i = 0; i < 27; i++) P[i] = (const float*)d_in[13 + i];
#define CWR(L) P[(L - 1) * 3 + 0]
#define CWN(L) P[(L - 1) * 3 + 1]
#define CB(L)  P[(L - 1) * 3 + 2]
    const float* fc1w = P[21]; const float* fc1b = P[22];
    const float* fc2w = P[23]; const float* fc2b = P[24];
    const float* fc3w = P[25]; const float* fc3b = P[26];

    float* ws = (float*)d_ws;
    const long long szH1 = (long long)cN1 * 64;
    const long long szT  = (long long)cN3 * 88;   // biggest temp: N3 x 88
    float* H1   = ws;
    float* TA   = H1 + szH1;
    float* TB   = TA + szT;
    float* AG   = TB + szT;
    float* CNT  = AG + szT;                       // cN3 floats
    float* XCAT = CNT + cN3;                      // G x 192
    float* GC   = XCAT + (long long)cG * 192;     // G
    float* Z1   = GC + cG;                        // G x 64
    float* Z2   = Z1 + (long long)cG * 64;        // G x 32
    float* LG   = Z2 + (long long)cG * 32;        // G x 10

    const int T = 256;
    auto zero = [&](float* p, long long nelem) {
        k_zero<<<cdiv_i(nelem, T), T, 0, stream>>>(p, nelem);
    };

    // zero pooled output accumulator once
    zero(XCAT, (long long)cG * 192);

    // =============== level 1 (nodes) ===============
    // c1: F0=16 -> 32
    zero(AG, (long long)cN1 * 16);
    k_scatter_edges<<<cdiv_i((long long)cE1 * 4, T), T, 0, stream>>>(x, ei1, AG, cE1, 4, 16);
    k_gconv_wmma<2><<<cdiv_i(cN1, 128), 256, 0, stream>>>(x, AG, CWR(1), CWN(1), CB(1), TA, cN1, 16, 1);
    // c2: 32 -> 64
    zero(AG, (long long)cN1 * 32);
    k_scatter_edges<<<cdiv_i((long long)cE1 * 8, T), T, 0, stream>>>(TA, ei1, AG, cE1, 8, 32);
    k_gconv_wmma<4><<<cdiv_i(cN1, 128), 256, 0, stream>>>(TA, AG, CWR(2), CWN(2), CB(2), TB, cN1, 32, 1);
    // c3: 64 -> 64 (result H1 persists for assignment pooling)
    zero(AG, (long long)cN1 * 64);
    k_scatter_edges<<<cdiv_i((long long)cE1 * 16, T), T, 0, stream>>>(TB, ei1, AG, cE1, 16, 64);
    k_gconv_wmma<4><<<cdiv_i(cN1, 128), 256, 0, stream>>>(TB, AG, CWR(3), CWN(3), CB(3), H1, cN1, 64, 1);
    // pool -> xcat[:, 0:64]
    zero(GC, cG);
    k_pool_scatter<<<cdiv_i((long long)cN1 * 16, T), T, 0, stream>>>(H1, b1, XCAT, GC, cN1, 0);
    k_pool_div<<<cdiv_i(cG * 64, T), T, 0, stream>>>(XCAT, GC, 0);

    // =============== level 2 (2-tuples) ===============
    zero(TB, (long long)cN2 * 64);
    zero(CNT, cN2);
    k_scatter_assign<<<cdiv_i((long long)cA2 * 16, T), T, 0, stream>>>(H1, as2s, as2d, TB, CNT, cA2);
    k_build_tuple<<<cdiv_i((long long)cN2 * 76, T), T, 0, stream>>>(TB, CNT, iso2, TA, cN2, cNI2);
    // c4: 76 -> 64
    zero(AG, (long long)cN2 * 76);
    k_scatter_edges<<<cdiv_i((long long)cE2 * 19, T), T, 0, stream>>>(TA, ei2, AG, cE2, 19, 76);
    k_gconv_wmma<4><<<cdiv_i(cN2, 128), 256, 0, stream>>>(TA, AG, CWR(4), CWN(4), CB(4), TB, cN2, 76, 1);
    // c5: 64 -> 64
    zero(AG, (long long)cN2 * 64);
    k_scatter_edges<<<cdiv_i((long long)cE2 * 16, T), T, 0, stream>>>(TB, ei2, AG, cE2, 16, 64);
    k_gconv_wmma<4><<<cdiv_i(cN2, 128), 256, 0, stream>>>(TB, AG, CWR(5), CWN(5), CB(5), TA, cN2, 64, 1);
    // pool -> xcat[:, 64:128]
    zero(GC, cG);
    k_pool_scatter<<<cdiv_i((long long)cN2 * 16, T), T, 0, stream>>>(TA, b2, XCAT, GC, cN2, 64);
    k_pool_div<<<cdiv_i(cG * 64, T), T, 0, stream>>>(XCAT, GC, 64);

    // =============== level 3 (3-tuples) ===============
    zero(TB, (long long)cN3 * 64);
    zero(CNT, cN3);
    k_scatter_assign<<<cdiv_i((long long)cA3 * 16, T), T, 0, stream>>>(H1, as3s, as3d, TB, CNT, cA3);
    k_build_tuple<<<cdiv_i((long long)cN3 * 88, T), T, 0, stream>>>(TB, CNT, iso3, TA, cN3, cNI3);
    // c6: 88 -> 64
    zero(AG, (long long)cN3 * 88);
    k_scatter_edges<<<cdiv_i((long long)cE3 * 22, T), T, 0, stream>>>(TA, ei3, AG, cE3, 22, 88);
    k_gconv_wmma<4><<<cdiv_i(cN3, 128), 256, 0, stream>>>(TA, AG, CWR(6), CWN(6), CB(6), TB, cN3, 88, 1);
    // c7: 64 -> 64
    zero(AG, (long long)cN3 * 64);
    k_scatter_edges<<<cdiv_i((long long)cE3 * 16, T), T, 0, stream>>>(TB, ei3, AG, cE3, 16, 64);
    k_gconv_wmma<4><<<cdiv_i(cN3, 128), 256, 0, stream>>>(TB, AG, CWR(7), CWN(7), CB(7), TA, cN3, 64, 1);
    // pool -> xcat[:, 128:192]
    zero(GC, cG);
    k_pool_scatter<<<cdiv_i((long long)cN3 * 16, T), T, 0, stream>>>(TA, b3, XCAT, GC, cN3, 128);
    k_pool_div<<<cdiv_i(cG * 64, T), T, 0, stream>>>(XCAT, GC, 128);

    // =============== head ===============
    k_fc<<<cdiv_i(cG * 64, T), T, 0, stream>>>(XCAT, fc1w, fc1b, Z1, cG, 192, 64, 1);
    k_fc<<<cdiv_i(cG * 32, T), T, 0, stream>>>(Z1, fc2w, fc2b, Z2, cG, 64, 32, 1);
    k_fc<<<cdiv_i(cG * cC, T), T, 0, stream>>>(Z2, fc3w, fc3b, LG, cG, 32, cC, 0);
    k_logsoftmax<<<cdiv_i(cG, T), T, 0, stream>>>(LG, (float*)d_out, cG);

    (void)in_sizes; (void)n_in; (void)out_size; (void)ws_size;
}
